// head_64716567216718
// MI455X (gfx1250) — matrix operand order repaired and checked by reading, not compiled, not run on hardware
//
#include <hip/hip_runtime.h>
#include <hip/hip_bf16.h>

typedef __attribute__((ext_vector_type(16))) _Float16 v16h;
typedef __attribute__((ext_vector_type(8)))  _Float16 v8h;
typedef __attribute__((ext_vector_type(8)))  float    v8f;
typedef __attribute__((ext_vector_type(4)))  float    v4f;

#define NB      512      // batch
#define NT      256      // tokens
#define NEMB    384      // embedding dim
#define NH      64       // head size
#define WT_STR  (NEMB + 8)   // padded Wt row stride (halves)
#define KK_STR  (NH + 8)     // padded kk row stride (halves) = 72
#define VT_STR  (NT + 8)     // padded v-transposed row stride (halves) = 264

// ---- fragment load helpers -------------------------------------------------
// A-fragment (16x32, 16-bit): per ISA table, lane half selects K-offset 0/8,
// elements 0..7 = K(base..base+7), elements 8..15 = K(16+base..16+base+7).
__device__ __forceinline__ v16h load_afrag(const _Float16* p) {
    v8h lo = *(const v8h*)(p);
    v8h hi = *(const v8h*)(p + 16);
    v16h r;
#pragma unroll
    for (int j = 0; j < 8; ++j) { r[j] = lo[j]; r[8 + j] = hi[j]; }
    return r;
}
// B-fragment (32x16, 16-bit): lanes 0-15 hold K=0..15, lanes 16-31 K=16..31,
// sequential within the lane (this code's contiguous order, not the ISA table order). Caller passes the
// lane-adjusted base pointer (already offset by 0 or 16 K).
__device__ __forceinline__ v16h load_bfrag(const _Float16* p) {
    v8h lo = *(const v8h*)(p);
    v8h hi = *(const v8h*)(p + 8);
    v16h r;
#pragma unroll
    for (int j = 0; j < 8; ++j) { r[j] = lo[j]; r[8 + j] = hi[j]; }
    return r;
}

// ---- Kernel 1: projection GEMM  out[f16] = scale * (X @ W + b) -------------
// X: [B*T, 384] f32, W: [384, 64] f32, out: [B*T, 64] f16.
// One workgroup = 8 waves * 16 rows = 128 rows. Wt staged transposed in LDS.
__global__ __launch_bounds__(256)
void proj_gemm(const float* __restrict__ X, const float* __restrict__ W,
               const float* __restrict__ bias, _Float16* __restrict__ out,
               float scale)
{
    __shared__ _Float16 wt[NH * WT_STR];   // 64 x 392 halves = 50,176 B

    const int tid = threadIdx.x;
    // W[k][n] (row-major, coalesced read) -> wt[n][k] (f16, padded rows)
    for (int idx = tid; idx < NEMB * NH; idx += 256) {
        int k = idx >> 6;        // / 64
        int n = idx & 63;
        wt[n * WT_STR + k] = (_Float16)W[idx];
    }
    __syncthreads();

    const int wave   = tid >> 5;
    const int lane   = tid & 31;
    const int laneN  = lane & 15;
    const int laneKA = (lane < 16) ? 0 : 8;    // A-fragment K interleave
    const int laneKB = (lane < 16) ? 0 : 16;   // B-fragment K split
    const int rowBase = blockIdx.x * 128 + wave * 16;

    v8f acc[4];
#pragma unroll
    for (int nt = 0; nt < 4; ++nt) {
        float bv = bias[nt * 16 + laneN];
#pragma unroll
        for (int j = 0; j < 8; ++j) acc[nt][j] = bv;
    }

    const float* xr = X + (size_t)(rowBase + laneN) * NEMB;

#pragma unroll 4
    for (int kt = 0; kt < NEMB / 32; ++kt) {
        const int k0 = kt * 32;
        const float* xp = xr + k0 + laneKA;
        v4f f0 = *(const v4f*)(xp);
        v4f f1 = *(const v4f*)(xp + 4);
        v4f f2 = *(const v4f*)(xp + 16);
        v4f f3 = *(const v4f*)(xp + 20);
        v16h a;
#pragma unroll
        for (int j = 0; j < 4; ++j) {
            a[j]      = (_Float16)f0[j];
            a[4 + j]  = (_Float16)f1[j];
            a[8 + j]  = (_Float16)f2[j];
            a[12 + j] = (_Float16)f3[j];
        }
#pragma unroll
        for (int nt = 0; nt < 4; ++nt) {
            const _Float16* wp = &wt[(nt * 16 + laneN) * WT_STR + k0 + laneKA];
            v16h b = load_afrag(wp);
            acc[nt] = __builtin_amdgcn_wmma_f32_16x16x32_f16(
                false, a, false, b, (short)0, acc[nt], false, false);
        }
    }

    const int mtop = laneKA;   // D: element j -> row (lane<16 ? j : j+8)
#pragma unroll
    for (int nt = 0; nt < 4; ++nt) {
#pragma unroll
        for (int j = 0; j < 8; ++j) {
            int row = rowBase + mtop + j;
            out[(size_t)row * NH + nt * 16 + laneN] =
                (_Float16)(acc[nt][j] * scale);
        }
    }
}

// ---- Kernel 2: causal flash attention, one workgroup per batch -------------
// Computes S^T = kk @ qq^T (so softmax stats are in-lane per query column),
// online softmax, O^T = V^T @ P^T. qq is pre-scaled by 384^-0.5.
// kk is staged into LDS with CDNA5 async global->LDS DMA (ASYNCcnt path).
__global__ __launch_bounds__(256)
void attn_head(const _Float16* __restrict__ qqg, const _Float16* __restrict__ kkg,
               const _Float16* __restrict__ vvg, float* __restrict__ out)
{
    extern __shared__ _Float16 smem[];
    _Float16* l_kk = smem;                    // [256][72] halves, LDS offset 0
    _Float16* l_vt = smem + NT * KK_STR;      // [64][264] halves (v transposed)

    const int b   = blockIdx.x;
    const int tid = threadIdx.x;
    const _Float16* qqb = qqg + (size_t)b * NT * NH;
    const _Float16* kkb = kkg + (size_t)b * NT * NH;
    const _Float16* vvb = vvg + (size_t)b * NT * NH;

    // Stage kk rows: async global->LDS b128 copies, no VGPR round-trip.
    // Dynamic LDS starts at offset 0 (no static LDS in this kernel), so the
    // LDS byte address of l_kk[row][c] is simply (row*KK_STR + c)*2.
    for (int idx = tid; idx < (NT * NH) / 8; idx += 256) {
        int row = idx >> 3;
        int c   = (idx & 7) * 8;
        unsigned int lds = (unsigned int)((row * KK_STR + c) * 2);
        unsigned long long ga =
            (unsigned long long)(uintptr_t)(kkb + row * NH + c);
        asm volatile("global_load_async_to_lds_b128 %0, %1, off"
                     :: "v"(lds), "v"(ga) : "memory");
    }
    // Stage v transposed: l_vt[h][t] (scatter, so regular path)
    for (int idx = tid; idx < (NT * NH) / 8; idx += 256) {
        int row = idx >> 3;
        int c   = (idx & 7) * 8;
        v8h d = *(const v8h*)(vvb + row * NH + c);
#pragma unroll
        for (int j = 0; j < 8; ++j) l_vt[(c + j) * VT_STR + row] = d[j];
    }
    asm volatile("s_wait_asynccnt 0x0" ::: "memory");  // async fills visible
    __syncthreads();

    const int wave   = tid >> 5;
    const int lane   = tid & 31;
    const int laneN  = lane & 15;
    const int laneKA = (lane < 16) ? 0 : 8;
    const int laneKB = (lane < 16) ? 0 : 16;
    const bool loHalf = (lane < 16);

    for (int pass = 0; pass < 2; ++pass) {
        const int qt = pass ? (15 - wave) : wave;   // query tile 0..15
        const int qb = qt * 16;
        const int q  = qb + laneN;

        // qq^T B-fragments for both K-steps of head dim (fixed per query tile)
        const _Float16* qrow = qqb + (size_t)(qb + laneN) * NH;
        v16h bq0 = load_afrag(qrow + laneKA);
        v16h bq1 = load_afrag(qrow + 32 + laneKA);

        v8f o[4] = {};                 // O^T accumulator: 64 h x 16 q
        float mrun = -__builtin_inff();
        float lrun = 0.0f;

        const int nPairs = (qt >> 1) + 1;   // 32-key pairs under the causal mask
        for (int p = 0; p < nPairs; ++p) {
            const int kb = p * 32;

            // S^T tiles for keys kb..kb+15 and kb+16..kb+31
            const _Float16* ar0 = l_kk + (kb + laneN) * KK_STR + laneKA;
            const _Float16* ar1 = ar0 + 16 * KK_STR;
            v8f s0 = {}, s1 = {};
            s0 = __builtin_amdgcn_wmma_f32_16x16x32_f16(
                false, load_afrag(ar0),      false, bq0, (short)0, s0, false, false);
            s0 = __builtin_amdgcn_wmma_f32_16x16x32_f16(
                false, load_afrag(ar0 + 32), false, bq1, (short)0, s0, false, false);
            s1 = __builtin_amdgcn_wmma_f32_16x16x32_f16(
                false, load_afrag(ar1),      false, bq0, (short)0, s1, false, false);
            s1 = __builtin_amdgcn_wmma_f32_16x16x32_f16(
                false, load_afrag(ar1 + 32), false, bq1, (short)0, s1, false, false);

            // causal mask + running max (combine lane pair l <-> l^16)
            float tmax = mrun;
#pragma unroll
            for (int j = 0; j < 8; ++j) {
                int key0 = kb + laneKA + j;
                float v0 = (key0 <= q)      ? s0[j] : -__builtin_inff();
                float v1 = (key0 + 16 <= q) ? s1[j] : -__builtin_inff();
                s0[j] = v0; s1[j] = v1;
                tmax = fmaxf(tmax, fmaxf(v0, v1));
            }
            tmax = fmaxf(tmax, __shfl_xor(tmax, 16, 32));
            const float corr = __expf(mrun - tmax);   // 0 on first tile
            mrun = tmax;

            // exp, row-sum, and pack P^T B-fragment (lane-pair exchange)
            float rs = 0.0f;
            v16h bp;
#pragma unroll
            for (int j = 0; j < 8; ++j) {
                float x0 = __expf(s0[j] - tmax);   // masked -> exp(-inf)=0
                float x1 = __expf(s1[j] - tmax);
                rs += x0 + x1;
                float p0 = __shfl_xor(x0, 16, 32);
                float p1 = __shfl_xor(x1, 16, 32);
                bp[j]     = (_Float16)(loHalf ? x0 : p1);
                bp[8 + j] = (_Float16)(loHalf ? p0 : x1);
            }
            rs += __shfl_xor(rs, 16, 32);
            lrun = lrun * corr + rs;
#pragma unroll
            for (int t = 0; t < 4; ++t)
#pragma unroll
                for (int j = 0; j < 8; ++j) o[t][j] *= corr;

            // O^T += V^T(16h x 32s) @ P^T(32s x 16q), 4 h-tiles
#pragma unroll
            for (int ht = 0; ht < 4; ++ht) {
                const _Float16* vr = l_vt + (ht * 16 + laneN) * VT_STR + kb + laneKB;
                o[ht] = __builtin_amdgcn_wmma_f32_16x16x32_f16(
                    false, load_bfrag(vr), false, bp, (short)0, o[ht], false, false);
            }
        }

        // epilogue: divide by row sum, store out[b][q][h] (two b128 per h-tile)
        const float inv = 1.0f / lrun;
        float* orow = out + ((size_t)b * NT + qb + laneN) * NH;
        const int mtop = laneKA;
#pragma unroll
        for (int ht = 0; ht < 4; ++ht) {
            v4f w0, w1;
#pragma unroll
            for (int j = 0; j < 4; ++j) {
                w0[j] = o[ht][j] * inv;
                w1[j] = o[ht][4 + j] * inv;
            }
            *(v4f*)(orow + ht * 16 + mtop)     = w0;
            *(v4f*)(orow + ht * 16 + mtop + 4) = w1;
        }
    }
}

// ---- launch ----------------------------------------------------------------
extern "C" void kernel_launch(void* const* d_in, const int* in_sizes, int n_in,
                              void* d_out, int out_size, void* d_ws, size_t ws_size,
                              hipStream_t stream) {
    const float* v  = (const float*)d_in[0];
    const float* k  = (const float*)d_in[1];
    const float* q  = (const float*)d_in[2];
    const float* Wq = (const float*)d_in[3];
    const float* bq = (const float*)d_in[4];
    const float* Wk = (const float*)d_in[5];
    const float* bk = (const float*)d_in[6];
    const float* Wv = (const float*)d_in[7];
    const float* bv = (const float*)d_in[8];
    float* out = (float*)d_out;

    const size_t projElems = (size_t)NB * NT * NH;   // 8.4M f16 each
    _Float16* qq = (_Float16*)d_ws;
    _Float16* kk = qq + projElems;
    _Float16* vv = kk + projElems;

    const float scale = 0.05103103630798287f;        // 384^-0.5 folded into qq
    const int gridProj = (NB * NT) / 128;            // 1024 workgroups

    proj_gemm<<<gridProj, 256, 0, stream>>>(q, Wq, bq, qq, scale);
    proj_gemm<<<gridProj, 256, 0, stream>>>(k, Wk, bk, kk, 1.0f);
    proj_gemm<<<gridProj, 256, 0, stream>>>(v, Wv, bv, vv, 1.0f);

    const size_t ldsBytes = (size_t)(NT * KK_STR + NH * VT_STR) * sizeof(_Float16); // 70,656 B
    attn_head<<<NB, 256, ldsBytes, stream>>>(qq, kk, vv, out);
}
